// XTransEncoderAct_42116449305114
// MI455X (gfx1250) — matrix operand order repaired and checked by reading, not compiled, not run on hardware
//
#include <hip/hip_runtime.h>
#include <hip/hip_bf16.h>

// ---------------- constants ----------------
#define BB   2
#define NN   2048
#define DD   256
#define NHH  8
#define LL   4
#define DFF_ 1024
#define DH_  32
#define D4_  1024
#define D2_  512
#define BN_  (BB*NN)        // 4096
#define BN3_ (BB*NN*3)      // 12288
#define FACTOR_ 0.08838834764831845f   // 0.5/sqrt(32)

typedef __attribute__((ext_vector_type(16))) _Float16 v16h;
typedef __attribute__((ext_vector_type(8)))  _Float16 v8h;
typedef __attribute__((ext_vector_type(8)))  float    v8f;

union U16 { v16h v; v8h h8[2]; };

__device__ __forceinline__ v8f wmma32(const U16& a, const U16& b, v8f c) {
  return __builtin_amdgcn_wmma_f32_16x16x32_f16(false, a.v, false, b.v, (short)0, c, false, false);
}

// ---------------- small elementwise kernels ----------------
__global__ void k_cvt_transpose(const float* __restrict__ W, _Float16* __restrict__ Wt,
                                int K, int Nc) {
  int i = blockIdx.x * 256 + threadIdx.x;
  if (i >= K * Nc) return;
  int k = i / Nc, n = i % Nc;
  Wt[(size_t)n * K + k] = (_Float16)W[i];
}

__global__ void k_cvt_f16(const float* __restrict__ X, _Float16* __restrict__ Y, int n) {
  int i = blockIdx.x * 256 + threadIdx.x;
  if (i < n) Y[i] = (_Float16)X[i];
}

// LayerNorm over D=256, one block (256 threads) per row, f16 output
__global__ __launch_bounds__(256) void k_layernorm(const float* __restrict__ X,
    const float* __restrict__ g, const float* __restrict__ bta,
    _Float16* __restrict__ Y) {
  const int row = blockIdx.x, tid = threadIdx.x;
  float x = X[(size_t)row * DD + tid];
  float s = x, ss = x * x;
  for (int off = 16; off; off >>= 1) {
    s  += __shfl_down(s,  off);
    ss += __shfl_down(ss, off);
  }
  __shared__ float sh[16];
  int w = tid >> 5, ln = tid & 31;
  if (ln == 0) { sh[w] = s; sh[8 + w] = ss; }
  __syncthreads();
  if (tid == 0) {
    float ts = 0.f, tss = 0.f;
    for (int i = 0; i < 8; ++i) { ts += sh[i]; tss += sh[8 + i]; }
    sh[0] = ts / (float)DD;
    sh[8] = tss / (float)DD;
  }
  __syncthreads();
  float mean = sh[0];
  float var  = sh[8] - mean * mean;
  float r = rsqrtf(var + 1e-5f);
  Y[(size_t)row * DD + tid] = (_Float16)((x - mean) * r * g[tid] + bta[tid]);
}

// Pack V_attn transposed: VaT[b,h,d,m], d in [0,128): d<32 -> Hv, else Vv
__global__ void k_pack_vat(const _Float16* __restrict__ Hv, const _Float16* __restrict__ Vv,
                           _Float16* __restrict__ VaT) {
  size_t i = (size_t)blockIdx.x * 256 + threadIdx.x;
  if (i >= (size_t)BB * NHH * 128 * NN) return;
  int m = (int)(i % NN);
  size_t t = i / NN;
  int d = (int)(t % 128);
  int bh = (int)(t / 128);
  int b = bh >> 3, h = bh & 7;
  _Float16 v;
  if (d < DH_) {
    v = Hv[((size_t)(b * NN + m)) * DD + h * DH_ + d];
  } else {
    int k3 = (d - DH_) >> 5, c = (d - DH_) & 31;
    v = Vv[(((size_t)(b * NN + m)) * 3 + k3) * DD + h * DH_ + c];
  }
  VaT[i] = v;
}

// scaler = concat(Hn2, ||V1||): V1 = Vp[..., :D]
__global__ void k_scaler(const _Float16* __restrict__ Hn2, const _Float16* __restrict__ Vp,
                         _Float16* __restrict__ scl) {
  int i = blockIdx.x * 256 + threadIdx.x;
  if (i >= BN_ * DD) return;
  int bn = i >> 8, d = i & 255;
  scl[(size_t)bn * D2_ + d] = Hn2[i];
  float s = 0.f;
  #pragma unroll
  for (int k = 0; k < 3; ++k) {
    float v = (float)Vp[(((size_t)bn * 3) + k) * D2_ + d];
    s += v * v;
  }
  scl[(size_t)bn * D2_ + DD + d] = (_Float16)sqrtf(s);
}

// H += sc[:, :D]; V += sc[:, D:, None] * V2 (V2 = Vp[..., D:])
__global__ void k_final(const float* __restrict__ sc, const _Float16* __restrict__ Vp,
                        float* __restrict__ Hout, float* __restrict__ Vout) {
  int i = blockIdx.x * 256 + threadIdx.x;
  if (i >= BN_ * DD) return;
  int bn = i >> 8, d = i & 255;
  Hout[i] += sc[(size_t)bn * D2_ + d];
  float s2 = sc[(size_t)bn * D2_ + DD + d];
  #pragma unroll
  for (int k = 0; k < 3; ++k) {
    size_t vi = ((size_t)bn * 3 + k) * DD + d;
    Vout[vi] += s2 * (float)Vp[((size_t)bn * 3 + k) * D2_ + DD + d];
  }
}

// ---------------- generic WMMA GEMM ----------------
// C[M,Nc] = A[M,K](f16) @ Bt[Nc,K]^T(f16) (+bias[col]) (+addsrc) (opt silu)
// block: 128 threads = 4 waves; per-wave tile 32x64 (2x4 WMMA tiles);
// block tile 128x64. A fragments reused 4x, B fragments 2x.
template<bool F16OUT, bool SILU>
__global__ __launch_bounds__(128) void k_gemm(const _Float16* __restrict__ A,
    const _Float16* __restrict__ Bt, const float* __restrict__ bias,
    const float* __restrict__ addsrc, void* __restrict__ outp,
    int M, int Nc, int K) {
  const int wave = threadIdx.x >> 5, lane = threadIdx.x & 31;
  const int hl = lane >> 4, lm = lane & 15;
  const int m0 = blockIdx.y * 128 + wave * 32;
  const int n0 = blockIdx.x * 64;
  // A frag: row = lm (+16 for second slab), k = hl*8+e / 16+hl*8+e
  const _Float16* ar0 = A + (size_t)(m0 + lm) * K + hl * 8;
  const _Float16* ar1 = ar0 + (size_t)16 * K;
  // B frag: col = lm, k = hl*8+e / 16+hl*8+e
  const _Float16* br[4];
  #pragma unroll
  for (int j = 0; j < 4; ++j)
    br[j] = Bt + (size_t)(n0 + j * 16 + lm) * K + hl * 8;

  v8f acc[2][4] = {};
  for (int k0 = 0; k0 < K; k0 += 32) {
    U16 a0, a1, b[4];
    a0.h8[0] = *(const v8h*)(ar0 + k0);
    a0.h8[1] = *(const v8h*)(ar0 + k0 + 16);
    a1.h8[0] = *(const v8h*)(ar1 + k0);
    a1.h8[1] = *(const v8h*)(ar1 + k0 + 16);
    #pragma unroll
    for (int j = 0; j < 4; ++j) {
      b[j].h8[0] = *(const v8h*)(br[j] + k0);
      b[j].h8[1] = *(const v8h*)(br[j] + k0 + 16);
    }
    #pragma unroll
    for (int j = 0; j < 4; ++j) {
      acc[0][j] = wmma32(a0, b[j], acc[0][j]);
      acc[1][j] = wmma32(a1, b[j], acc[1][j]);
    }
  }
  #pragma unroll
  for (int i = 0; i < 2; ++i) {
    const int mb = m0 + i * 16 + hl * 8;
    #pragma unroll
    for (int j = 0; j < 4; ++j) {
      const int col = n0 + j * 16 + lm;
      const float bc = bias ? bias[col] : 0.0f;
      #pragma unroll
      for (int r = 0; r < 8; ++r) {
        float v = acc[i][j][r] + bc;
        if (addsrc) v += addsrc[(size_t)(mb + r) * Nc + col];
        if (SILU)   v = v / (1.0f + __expf(-v));
        if (F16OUT) ((_Float16*)outp)[(size_t)(mb + r) * Nc + col] = (_Float16)v;
        else        ((float*)outp)[(size_t)(mb + r) * Nc + col] = v;
      }
    }
  }
}

// ---------------- flash attention ----------------
// one wave per 16-row query tile of one (b,h); 8 waves/block
__global__ __launch_bounds__(256) void k_flash(const _Float16* __restrict__ Q,
    const _Float16* __restrict__ Km, const _Float16* __restrict__ VaT,
    const float* __restrict__ rbf, const float* __restrict__ Db,
    const unsigned char* __restrict__ mask,
    _Float16* __restrict__ Hres, _Float16* __restrict__ Vres) {
  __shared__ __align__(16) _Float16 pst[8][16][32];
  const int wave = threadIdx.x >> 5, lane = threadIdx.x & 31;
  const int hl = lane >> 4, lm = lane & 15;
  const int task = blockIdx.x * 8 + wave;
  const int qt = task & (NN / 16 - 1);
  const int bh = task >> 7;
  const int b = bh >> 3, h = bh & 7;
  const int n0 = qt * 16;

  // preload Q fragments for this query tile (K-dim 128 -> 4 frags)
  U16 qf[4];
  {
    const _Float16* qb = Q + ((size_t)(b * NN + n0 + lm)) * D4_ + h * 128 + hl * 8;
    #pragma unroll
    for (int kk = 0; kk < 4; ++kk) {
      qf[kk].h8[0] = *(const v8h*)(qb + kk * 32);
      qf[kk].h8[1] = *(const v8h*)(qb + kk * 32 + 16);
    }
  }
  float mi[8], li[8];
  #pragma unroll
  for (int r = 0; r < 8; ++r) { mi[r] = -3.0e38f; li[r] = 0.0f; }
  v8f oa[8] = {};

  const float* rbf_b = rbf + (size_t)b * NN * NN;
  const float* db_b  = Db  + (size_t)b * NN * NN;

  for (int m0 = 0; m0 < NN; m0 += 32) {
    float p[2][8];
    // S = Q @ K^T for two 16-key tiles, + bias, * FACTOR
    #pragma unroll
    for (int t = 0; t < 2; ++t) {
      const _Float16* kb = Km + ((size_t)(b * NN + m0 + t * 16 + lm)) * D4_ + h * 128 + hl * 8;
      v8f s = {};
      #pragma unroll
      for (int kk = 0; kk < 4; ++kk) {
        U16 bf;
        bf.h8[0] = *(const v8h*)(kb + kk * 32);
        bf.h8[1] = *(const v8h*)(kb + kk * 32 + 16);
        s = wmma32(qf[kk], bf, s);
      }
      const int mc = m0 + t * 16 + lm;
      const bool mk = mask[b * NN + mc] != 0;
      #pragma unroll
      for (int r = 0; r < 8; ++r) {
        const int rr = n0 + hl * 8 + r;
        float bias = mk ? (rbf_b[(size_t)rr * NN + mc] + db_b[(size_t)rr * NN + mc]) : -3.0e38f;
        p[t][r] = s[r] * FACTOR_ + bias;
      }
    }
    // online softmax update (row reductions across the 16-lane half-groups)
    float sc[8];
    #pragma unroll
    for (int r = 0; r < 8; ++r) {
      float v = fmaxf(p[0][r], p[1][r]);
      v = fmaxf(v, __shfl_xor(v, 1));
      v = fmaxf(v, __shfl_xor(v, 2));
      v = fmaxf(v, __shfl_xor(v, 4));
      v = fmaxf(v, __shfl_xor(v, 8));
      float mn = fmaxf(mi[r], v);
      sc[r] = __expf(mi[r] - mn);
      p[0][r] = __expf(p[0][r] - mn);
      p[1][r] = __expf(p[1][r] - mn);
      float rs = p[0][r] + p[1][r];
      rs += __shfl_xor(rs, 1);
      rs += __shfl_xor(rs, 2);
      rs += __shfl_xor(rs, 4);
      rs += __shfl_xor(rs, 8);
      li[r] = li[r] * sc[r] + rs;
      mi[r] = mn;
    }
    // rescale output accumulators
    #pragma unroll
    for (int dt = 0; dt < 8; ++dt)
      #pragma unroll
      for (int r = 0; r < 8; ++r) oa[dt][r] *= sc[r];
    // P: C-layout -> A-layout via per-wave LDS tile
    #pragma unroll
    for (int t = 0; t < 2; ++t)
      #pragma unroll
      for (int r = 0; r < 8; ++r)
        pst[wave][hl * 8 + r][t * 16 + lm] = (_Float16)p[t][r];
    asm volatile("s_wait_dscnt 0" ::: "memory");
    U16 pf;
    pf.h8[0] = *(const v8h*)&pst[wave][lm][hl * 8];
    pf.h8[1] = *(const v8h*)&pst[wave][lm][16 + hl * 8];
    // O += P @ V_attn  (8 column tiles of 16 over the 128-wide value)
    #pragma unroll
    for (int dt = 0; dt < 8; ++dt) {
      const _Float16* vb = VaT + ((size_t)((b * NHH + h) * 128 + dt * 16 + lm)) * NN + m0 + hl * 8;
      U16 bf;
      bf.h8[0] = *(const v8h*)(vb);
      bf.h8[1] = *(const v8h*)(vb + 16);
      oa[dt] = wmma32(pf, bf, oa[dt]);
    }
  }
  // epilogue: normalize + scatter to Hres / Vres (f16)
  #pragma unroll
  for (int dt = 0; dt < 8; ++dt) {
    const int d = dt * 16 + lm;
    #pragma unroll
    for (int r = 0; r < 8; ++r) {
      const int nr = n0 + hl * 8 + r;
      float v = oa[dt][r] / li[r];
      if (d < DH_) {
        Hres[((size_t)(b * NN + nr)) * DD + h * DH_ + d] = (_Float16)v;
      } else {
        const int k3 = (d - DH_) >> 5, c = (d - DH_) & 31;
        Vres[(((size_t)(b * NN + nr)) * 3 + k3) * DD + h * DH_ + c] = (_Float16)v;
      }
    }
  }
}

// ---------------- host ----------------
static inline dim3 gemm_grid(int M, int Nc) { return dim3(Nc / 64, M / 128); }

extern "C" void kernel_launch(void* const* d_in, const int* in_sizes, int n_in,
                              void* d_out, int out_size, void* d_ws, size_t ws_size,
                              hipStream_t stream) {
  (void)in_sizes; (void)n_in; (void)out_size; (void)ws_size;
  const float* H_in  = (const float*)d_in[0];
  const float* V_in  = (const float*)d_in[1];
  const float* Db    = (const float*)d_in[2];
  const float* rbf   = (const float*)d_in[3];
  const unsigned char* mask = (const unsigned char*)d_in[4];
  const float* ln1_g = (const float*)d_in[5];
  const float* ln1_b = (const float*)d_in[6];
  const float* ln2_g = (const float*)d_in[7];
  const float* ln2_b = (const float*)d_in[8];
  const float* Wq  = (const float*)d_in[9];
  const float* bq  = (const float*)d_in[10];
  const float* Wk  = (const float*)d_in[11];
  const float* bk  = (const float*)d_in[12];
  const float* Wv  = (const float*)d_in[13];
  const float* bv  = (const float*)d_in[14];
  const float* Wvv = (const float*)d_in[15];
  const float* Wo  = (const float*)d_in[16];
  const float* bo  = (const float*)d_in[17];
  const float* Wvo = (const float*)d_in[18];
  const float* Wlv = (const float*)d_in[19];
  const float* W1  = (const float*)d_in[20];
  const float* b1  = (const float*)d_in[21];
  const float* W2  = (const float*)d_in[22];
  const float* b2  = (const float*)d_in[23];

  float* Hout = (float*)d_out;
  float* Vout = Hout + (size_t)BN_ * DD;

  // init residual state in d_out
  hipMemcpyAsync(Hout, H_in, (size_t)BN_ * DD * sizeof(float), hipMemcpyDeviceToDevice, stream);
  hipMemcpyAsync(Vout, V_in, (size_t)BN3_ * DD * sizeof(float), hipMemcpyDeviceToDevice, stream);

  // workspace layout
  char* w = (char*)d_ws;
  size_t off = 0;
  auto alloc = [&](size_t bytes) -> char* {
    char* p = w + off;
    off = (off + bytes + 255) & ~(size_t)255;
    return p;
  };
  _Float16* Hn16   = (_Float16*)alloc((size_t)BN_ * DD * 2);
  _Float16* V16    = (_Float16*)alloc((size_t)BN3_ * DD * 2);
  _Float16* Q16    = (_Float16*)alloc((size_t)BN_ * D4_ * 2);
  _Float16* K16    = (_Float16*)alloc((size_t)BN_ * D4_ * 2);
  _Float16* Hv16   = (_Float16*)alloc((size_t)BN_ * DD * 2);
  _Float16* Vv16   = (_Float16*)alloc((size_t)BN3_ * DD * 2);
  _Float16* VaT16  = (_Float16*)alloc((size_t)BB * NHH * 128 * NN * 2);
  _Float16* Hres16 = (_Float16*)alloc((size_t)BN_ * DD * 2);
  _Float16* Vres16 = (_Float16*)alloc((size_t)BN3_ * DD * 2);
  _Float16* Vp16   = (_Float16*)alloc((size_t)BN3_ * D2_ * 2);
  _Float16* scl16  = (_Float16*)alloc((size_t)BN_ * D2_ * 2);
  _Float16* h116   = (_Float16*)alloc((size_t)BN_ * DFF_ * 2);
  float*    sc32   = (float*)   alloc((size_t)BN_ * D2_ * 4);
  _Float16* WqT  = (_Float16*)alloc((size_t)D4_ * DD * 2);
  _Float16* WkT  = (_Float16*)alloc((size_t)D4_ * DD * 2);
  _Float16* WvT  = (_Float16*)alloc((size_t)DD * DD * 2);
  _Float16* WvvT = (_Float16*)alloc((size_t)DD * DD * 2);
  _Float16* WoT  = (_Float16*)alloc((size_t)DD * DD * 2);
  _Float16* WvoT = (_Float16*)alloc((size_t)DD * DD * 2);
  _Float16* WlvT = (_Float16*)alloc((size_t)D2_ * DD * 2);
  _Float16* W1T  = (_Float16*)alloc((size_t)DFF_ * D2_ * 2);
  _Float16* W2T  = (_Float16*)alloc((size_t)D2_ * DFF_ * 2);

  auto cvtT = [&](const float* src, _Float16* dst, int K, int Nc) {
    int n = K * Nc;
    k_cvt_transpose<<<(n + 255) / 256, 256, 0, stream>>>(src, dst, K, Nc);
  };

  for (int l = 0; l < LL; ++l) {
    // ---- weight convert+transpose (f32 -> f16, [K,Nc] -> [Nc,K]) ----
    cvtT(Wq  + (size_t)l * DD * D4_,  WqT,  DD, D4_);
    cvtT(Wk  + (size_t)l * DD * D4_,  WkT,  DD, D4_);
    cvtT(Wv  + (size_t)l * DD * DD,   WvT,  DD, DD);
    cvtT(Wvv + (size_t)l * DD * DD,   WvvT, DD, DD);
    cvtT(Wo  + (size_t)l * DD * DD,   WoT,  DD, DD);
    cvtT(Wvo + (size_t)l * DD * DD,   WvoT, DD, DD);
    cvtT(Wlv + (size_t)l * DD * D2_,  WlvT, DD, D2_);
    cvtT(W1  + (size_t)l * D2_ * DFF_, W1T, D2_, DFF_);
    cvtT(W2  + (size_t)l * DFF_ * D2_, W2T, DFF_, D2_);

    // ---- attention half ----
    k_layernorm<<<BN_, 256, 0, stream>>>(Hout, ln1_g + l * DD, ln1_b + l * DD, Hn16);
    k_gemm<true, false><<<gemm_grid(BN_, D4_), 128, 0, stream>>>(
        Hn16, WqT, bq + (size_t)l * D4_, nullptr, Q16, BN_, D4_, DD);
    k_gemm<true, false><<<gemm_grid(BN_, D4_), 128, 0, stream>>>(
        Hn16, WkT, bk + (size_t)l * D4_, nullptr, K16, BN_, D4_, DD);
    k_gemm<true, false><<<gemm_grid(BN_, DD), 128, 0, stream>>>(
        Hn16, WvT, bv + (size_t)l * DD, nullptr, Hv16, BN_, DD, DD);
    k_cvt_f16<<<(BN3_ * DD + 255) / 256, 256, 0, stream>>>(Vout, V16, BN3_ * DD);
    k_gemm<true, false><<<gemm_grid(BN3_, DD), 128, 0, stream>>>(
        V16, WvvT, nullptr, nullptr, Vv16, BN3_, DD, DD);
    {
      size_t n = (size_t)BB * NHH * 128 * NN;
      k_pack_vat<<<(unsigned)((n + 255) / 256), 256, 0, stream>>>(Hv16, Vv16, VaT16);
    }
    k_flash<<<BB * NHH * (NN / 16) / 8, 256, 0, stream>>>(
        Q16, K16, VaT16, rbf + (size_t)l * BB * NN * NN, Db, mask, Hres16, Vres16);
    // H += Hres @ Wo + bo ; V += Vres @ Wvo
    k_gemm<false, false><<<gemm_grid(BN_, DD), 128, 0, stream>>>(
        Hres16, WoT, bo + (size_t)l * DD, Hout, Hout, BN_, DD, DD);
    k_gemm<false, false><<<gemm_grid(BN3_, DD), 128, 0, stream>>>(
        Vres16, WvoT, nullptr, Vout, Vout, BN3_, DD, DD);

    // ---- MLP half ----
    k_layernorm<<<BN_, 256, 0, stream>>>(Hout, ln2_g + l * DD, ln2_b + l * DD, Hn16);
    k_cvt_f16<<<(BN3_ * DD + 255) / 256, 256, 0, stream>>>(Vout, V16, BN3_ * DD);
    k_gemm<true, false><<<gemm_grid(BN3_, D2_), 128, 0, stream>>>(
        V16, WlvT, nullptr, nullptr, Vp16, BN3_, D2_, DD);
    k_scaler<<<(BN_ * DD + 255) / 256, 256, 0, stream>>>(Hn16, Vp16, scl16);
    k_gemm<true, true><<<gemm_grid(BN_, DFF_), 128, 0, stream>>>(
        scl16, W1T, b1 + (size_t)l * DFF_, nullptr, h116, BN_, DFF_, D2_);
    k_gemm<false, false><<<gemm_grid(BN_, D2_), 128, 0, stream>>>(
        h116, W2T, b2 + (size_t)l * D2_, nullptr, sc32, BN_, D2_, DFF_);
    k_final<<<(BN_ * DD + 255) / 256, 256, 0, stream>>>(sc32, Vp16, Hout, Vout);
  }
}
